// Attention_Attention_85272280695427
// MI455X (gfx1250) — matrix operand order repaired and checked by reading, not compiled, not run on hardware
//
#include <hip/hip_runtime.h>

// ---------------------------------------------------------------------------
// Types for CDNA5 WMMA (wave32, v_wmma_f32_16x16x32_bf16)
// ---------------------------------------------------------------------------
typedef __attribute__((ext_vector_type(16))) __bf16 v16bf;
typedef __attribute__((ext_vector_type(8)))  __bf16 v8bf;
typedef __attribute__((ext_vector_type(8)))  float  v8f;

union FragA { v16bf v; v8bf h[2]; };

__device__ __forceinline__ unsigned short f2bf(float f) {
  unsigned int u = __float_as_uint(f);
  u += 0x7FFFu + ((u >> 16) & 1u);      // round-to-nearest-even
  return (unsigned short)(u >> 16);
}
__device__ __forceinline__ float bf2f(unsigned short v) {
  return __uint_as_float(((unsigned int)v) << 16);
}

// ---------------------------------------------------------------------------
// Canonical batched GEMM: C[M,N] = epi( A[M,K] @ Bt[N,K]^T + bias )
//   A, Bt: bf16 row-major (K contiguous).  BIAS_MODE: 0 none, 1 by M, 2 by N.
//   SILU: apply x*sigmoid(x).  OUTF32: write f32 (else bf16).
//   Block = 128 threads = 4 waves; wave tile 64x64 (4x4 WMMA); block 128x128.
//   Requires M%128==0, N%128==0, K%32==0 (true for all shapes here).
// ---------------------------------------------------------------------------
template<int BIAS_MODE, int SILU, int OUTF32>
__device__ __forceinline__ void epi_store(v8f c, void* __restrict__ Cout,
                                          long long cbase, const float* __restrict__ bias,
                                          int mrow, int ncol, int ldc, int hf) {
#pragma unroll
  for (int rr = 0; rr < 8; ++rr) {
    int row = mrow + rr + 8 * hf;
    float v = c[rr];
    if constexpr (BIAS_MODE == 1) v += bias[row];
    if constexpr (BIAS_MODE == 2) v += bias[ncol];
    if constexpr (SILU != 0)      v = v / (1.0f + __expf(-v));
    long long o = cbase + (long long)row * ldc + ncol;
    if constexpr (OUTF32 != 0) ((float*)Cout)[o] = v;
    else                       ((unsigned short*)Cout)[o] = f2bf(v);
  }
}

template<int BIAS_MODE, int SILU, int OUTF32>
__global__ __launch_bounds__(128) void gemm_bf16(
    const unsigned short* __restrict__ A,
    const unsigned short* __restrict__ Bt,
    const float* __restrict__ bias,
    void* __restrict__ Cout,
    int K, int lda, int ldb, int ldc,
    long long sA, long long sB, long long sC)
{
  const int z = blockIdx.z;
  const unsigned short* Ab = A  + (long long)z * sA;
  const unsigned short* Bb = Bt + (long long)z * sB;
  const long long cbase = (long long)z * sC;

  const int wave = threadIdx.x >> 5;      // wave32: 0..3
  const int lane = threadIdx.x & 31;
  const int wm = wave & 1;                // 2 waves along M
  const int wn = wave >> 1;               // 2 waves along N
  const int m0 = blockIdx.x * 128 + wm * 64;
  const int n0 = blockIdx.y * 128 + wn * 64;
  const int r  = lane & 15;
  const int hf = lane >> 4;

  // A fragment: lane holds row m+r, K = hf*16..+16 (contiguous, matches B)
  // B fragment: lane holds column n+r, K = hf*16..+16 (contiguous 32B)
  const unsigned short* ap[4];
  const unsigned short* bp[4];
#pragma unroll
  for (int t = 0; t < 4; ++t) {
    ap[t] = Ab + (long long)(m0 + t * 16 + r) * lda + hf * 16;
    bp[t] = Bb + (long long)(n0 + t * 16 + r) * ldb + hf * 16;
  }

  v8f c[4][4] = {};
  for (int k = 0; k < K; k += 32) {
    FragA a[4];
    v16bf b[4];
#pragma unroll
    for (int t = 0; t < 4; ++t) {
      a[t].h[0] = *(const v8bf*)(ap[t] + k);
      a[t].h[1] = *(const v8bf*)(ap[t] + k + 8);
    }
#pragma unroll
    for (int t = 0; t < 4; ++t) b[t] = *(const v16bf*)(bp[t] + k);
#pragma unroll
    for (int i = 0; i < 4; ++i)
#pragma unroll
      for (int j = 0; j < 4; ++j)
        c[i][j] = __builtin_amdgcn_wmma_f32_16x16x32_bf16(
            false, a[i].v, false, b[j], (short)0, c[i][j], false, false);
  }

#pragma unroll
  for (int i = 0; i < 4; ++i)
#pragma unroll
    for (int j = 0; j < 4; ++j)
      epi_store<BIAS_MODE,SILU,OUTF32>(c[i][j], Cout, cbase, bias,
                                       m0 + i * 16, n0 + j * 16 + r, ldc, hf);
}

// ---------------------------------------------------------------------------
// Row softmax: f32 in -> bf16 out. One block (256 thr) per row, rowlen <= 768.
// ---------------------------------------------------------------------------
__global__ __launch_bounds__(256) void softmax_rows(
    const float* __restrict__ S, unsigned short* __restrict__ O, int n)
{
  __shared__ float red[256];
  __shared__ float buf[768];
  const long long row = blockIdx.x;
  const float* p = S + row * n;
  const int t = threadIdx.x;

  float mx = -3.4e38f;
  for (int i = t; i < n; i += 256) { float v = p[i]; buf[i] = v; mx = fmaxf(mx, v); }
  red[t] = mx; __syncthreads();
  for (int s = 128; s > 0; s >>= 1) { if (t < s) red[t] = fmaxf(red[t], red[t + s]); __syncthreads(); }
  mx = red[0]; __syncthreads();

  float sum = 0.f;
  for (int i = t; i < n; i += 256) { float e = __expf(buf[i] - mx); buf[i] = e; sum += e; }
  red[t] = sum; __syncthreads();
  for (int s = 128; s > 0; s >>= 1) { if (t < s) red[t] += red[t + s]; __syncthreads(); }
  const float inv = 1.0f / red[0];

  unsigned short* q = O + row * n;
  for (int i = t; i < n; i += 256) q[i] = f2bf(buf[i] * inv);
}

// f32 -> bf16 convert
__global__ __launch_bounds__(256) void cvt_f32_bf16(
    const float* __restrict__ s, unsigned short* __restrict__ d, int n)
{
  int i = blockIdx.x * 256 + threadIdx.x;
  if (i < n) d[i] = f2bf(s[i]);
}

// pooled[b,h] = mean_n h[b,n,h] ; h: [256,256,384] bf16 -> pooled [256,384] bf16
__global__ __launch_bounds__(256) void pool_mean(
    const unsigned short* __restrict__ h, unsigned short* __restrict__ p)
{
  int idx = blockIdx.x * 256 + threadIdx.x;   // 0 .. 256*384
  int b = idx / 384, hh = idx - b * 384;
  const unsigned short* q = h + (long long)b * 256 * 384 + hh;
  float s = 0.f;
  for (int n = 0; n < 256; ++n) s += bf2f(q[(long long)n * 384]);
  p[idx] = f2bf(s * (1.0f / 256.0f));
}

// ---------------------------------------------------------------------------
// Host orchestration
// ---------------------------------------------------------------------------
extern "C" void kernel_launch(void* const* d_in, const int* in_sizes, int n_in,
                              void* d_out, int out_size, void* d_ws, size_t ws_size,
                              hipStream_t stream) {
  (void)in_sizes; (void)n_in; (void)out_size; (void)ws_size;
  const float* x   = (const float*)d_in[0];
  const float* Wi  = (const float*)d_in[1];
  const float* bi  = (const float*)d_in[2];
  const float* Aq4 = (const float*)d_in[3];
  const float* Ak4 = (const float*)d_in[4];
  const float* Av4 = (const float*)d_in[5];
  const float* Bq4 = (const float*)d_in[6];
  const float* Bk4 = (const float*)d_in[7];
  const float* Bv4 = (const float*)d_in[8];
  const float* W4  = (const float*)d_in[9];
  const float* b4  = (const float*)d_in[10];
  const float* Aq7 = (const float*)d_in[11];
  const float* Ak7 = (const float*)d_in[12];
  const float* Av7 = (const float*)d_in[13];
  const float* Bq7 = (const float*)d_in[14];
  const float* Bk7 = (const float*)d_in[15];
  const float* Bv7 = (const float*)d_in[16];
  const float* Wo  = (const float*)d_in[17];
  const float* bo  = (const float*)d_in[18];

  char* w = (char*)d_ws;
  size_t off = 0;
  auto take = [&](size_t elems, size_t esz) {
    size_t r = off;
    off = (off + elems * esz + 255) & ~(size_t)255;
    return r;
  };
  // bf16 copies of inputs / weights
  unsigned short* xbf  = (unsigned short*)(w + take(16777216, 2));        // [B,NA,256]
  unsigned short* wWi  = (unsigned short*)(w + take(768 * 256, 2));
  unsigned short* wAq4 = (unsigned short*)(w + take(768 * 768, 2));
  unsigned short* wAk4 = (unsigned short*)(w + take(768 * 768, 2));
  unsigned short* wAv4 = (unsigned short*)(w + take(768 * 768, 2));
  unsigned short* wW4  = (unsigned short*)(w + take(384 * 768, 2));
  unsigned short* wAq7 = (unsigned short*)(w + take(384 * 384, 2));
  unsigned short* wAk7 = (unsigned short*)(w + take(384 * 384, 2));
  unsigned short* wAv7 = (unsigned short*)(w + take(384 * 384, 2));
  unsigned short* wWo  = (unsigned short*)(w + take(128 * 384, 2));
  // activation buffers [B*NA*768] bf16, reused across stages
  const long long EBUF = 256LL * 256 * 768;                                // 50,331,648
  unsigned short* buf0 = (unsigned short*)(w + take((size_t)EBUF, 2));
  unsigned short* buf1 = (unsigned short*)(w + take((size_t)EBUF, 2));
  unsigned short* buf2 = (unsigned short*)(w + take((size_t)EBUF, 2));
  unsigned short* buf3 = (unsigned short*)(w + take((size_t)EBUF, 2));
  // chunked score buffers (32 batches)
  float*          Sf32 = (float*)(w + take(32LL * 768 * 768, 4));
  unsigned short* Sbf  = (unsigned short*)(w + take(32LL * 768 * 768, 2));
  unsigned short* pooled = (unsigned short*)(w + take(256 * 384, 2));

  dim3 blk(128);
  dim3 blk256(256);
  auto cvt = [&](const float* s, unsigned short* d, int n) {
    cvt_f32_bf16<<<(n + 255) / 256, blk256, 0, stream>>>(s, d, n);
  };
  cvt(x,   xbf,  16777216);
  cvt(Wi,  wWi,  768 * 256);
  cvt(Aq4, wAq4, 768 * 768);  cvt(Ak4, wAk4, 768 * 768);  cvt(Av4, wAv4, 768 * 768);
  cvt(W4,  wW4,  384 * 768);
  cvt(Aq7, wAq7, 384 * 384);  cvt(Ak7, wAk7, 384 * 384);  cvt(Av7, wAv7, 384 * 384);
  cvt(Wo,  wWo,  128 * 384);

  // --- mlp_in: h1[b][n,d] = silu(x[b] @ Wi^T + bi), [NA,768] ---
  gemm_bf16<2,1,0><<<dim3(2,6,256), blk, 0, stream>>>(
      xbf, wWi, bi, buf0, 256, 256, 256, 768, 65536LL, 0LL, 196608LL);

  // --- block4 projections (D=768) ---
  // Q4[b][d,n] = silu(Aq4 @ h1[b]^T + Bq4)   [768,256]
  gemm_bf16<1,1,0><<<dim3(6,2,256), blk, 0, stream>>>(
      wAq4, buf0, Bq4, buf1, 768, 768, 768, 256, 0LL, 196608LL, 196608LL);
  gemm_bf16<1,1,0><<<dim3(6,2,256), blk, 0, stream>>>(
      wAk4, buf0, Bk4, buf2, 768, 768, 768, 256, 0LL, 196608LL, 196608LL);
  // V4t[b][n,d] = silu(h1[b] @ Av4^T + Bv4)  [256,768]
  gemm_bf16<2,1,0><<<dim3(2,6,256), blk, 0, stream>>>(
      buf0, wAv4, Bv4, buf3, 768, 768, 768, 768, 196608LL, 0LL, 196608LL);

  // --- block4 attention, 8 chunks of 32 batches ---
  for (int c = 0; c < 8; ++c) {
    long long o = (long long)c * 32 * 196608;
    // S[d,e] = Q[b] @ K[b]^T  (f32, no epilogue)
    gemm_bf16<0,0,1><<<dim3(6,6,32), blk, 0, stream>>>(
        buf1 + o, buf2 + o, nullptr, Sf32, 256, 256, 256, 768,
        196608LL, 196608LL, 589824LL);
    softmax_rows<<<32 * 768, blk256, 0, stream>>>(Sf32, Sbf, 768);
    // out[n,d] = silu( sum_e V4t[n,e] * S[d,e] ) -> overwrite h1 region
    gemm_bf16<0,1,0><<<dim3(2,6,32), blk, 0, stream>>>(
        buf3 + o, Sbf, nullptr, (void*)(buf0 + o), 768, 768, 768, 768,
        196608LL, 589824LL, 196608LL);
  }

  // --- mlp_hidden_4: h3[b][n,h] = silu(attn4[b] @ W4^T + b4)  [256,384] ---
  gemm_bf16<2,1,0><<<dim3(2,3,256), blk, 0, stream>>>(
      buf0, wW4, b4, buf1, 768, 768, 768, 384, 196608LL, 0LL, 98304LL);

  // --- block7 projections (D=384) ---
  gemm_bf16<1,1,0><<<dim3(3,2,256), blk, 0, stream>>>(
      wAq7, buf1, Bq7, buf2, 384, 384, 384, 256, 0LL, 98304LL, 98304LL);
  gemm_bf16<1,1,0><<<dim3(3,2,256), blk, 0, stream>>>(
      wAk7, buf1, Bk7, buf3, 384, 384, 384, 256, 0LL, 98304LL, 98304LL);
  gemm_bf16<2,1,0><<<dim3(2,3,256), blk, 0, stream>>>(
      buf1, wAv7, Bv7, buf0, 384, 384, 384, 384, 98304LL, 0LL, 98304LL);

  // --- block7 attention, 8 chunks of 32 batches ---
  for (int c = 0; c < 8; ++c) {
    long long o = (long long)c * 32 * 98304;
    gemm_bf16<0,0,1><<<dim3(3,3,32), blk, 0, stream>>>(
        buf2 + o, buf3 + o, nullptr, Sf32, 256, 256, 256, 384,
        98304LL, 98304LL, 147456LL);
    softmax_rows<<<32 * 384, blk256, 0, stream>>>(Sf32, Sbf, 384);
    gemm_bf16<0,1,0><<<dim3(2,3,32), blk, 0, stream>>>(
        buf0 + o, Sbf, nullptr, (void*)(buf1 + o), 384, 384, 384, 384,
        98304LL, 147456LL, 98304LL);
  }

  // --- mean over agents -> pooled [256,384] bf16 ---
  pool_mean<<<384, blk256, 0, stream>>>(buf1, pooled);

  // --- mlp_out: out[b,o] = silu(pooled @ Wo^T + bo) -> f32 d_out [256,128] ---
  gemm_bf16<2,1,1><<<dim3(2,1,1), blk, 0, stream>>>(
      pooled, wWo, bo, d_out, 384, 384, 384, 128, 0LL, 0LL, 0LL);
}
